// AdditiveAttention_35029753266750
// MI455X (gfx1250) — matrix operand order repaired and checked by reading, not compiled, not run on hardware
//
#include <hip/hip_runtime.h>
#include <hip/hip_bf16.h>
#include <math.h>

typedef __attribute__((ext_vector_type(16))) __bf16 v16bf;
typedef __attribute__((ext_vector_type(8)))  float  v8f;

#define T_DIM 2048
#define N_DIM 64
#define E_DIM 1024
#define A_DIM 256
#define M_TOT (T_DIM * N_DIM)   // 131072 GEMM rows

// ---- CDNA5 async global->LDS path (ASYNCcnt-tracked), with fallback ----
#if defined(__has_builtin)
#  if __has_builtin(__builtin_amdgcn_global_load_async_to_lds_b128)
#    define HAVE_ASYNC 1
#  endif
#endif
#ifndef HAVE_ASYNC
#  define HAVE_ASYNC 0
#endif

#if HAVE_ASYNC
#  if __has_builtin(__builtin_amdgcn_s_wait_asynccnt)
#    define WAIT_ASYNC() __builtin_amdgcn_s_wait_asynccnt(0)
#  else
#    define WAIT_ASYNC() asm volatile("s_wait_asynccnt 0" ::: "memory")
#  endif
#else
#  define WAIT_ASYNC() ((void)0)
#endif

// builtin prototype (from hipcc diagnostic): param0 = global ("__device__")
// pointer to int __attribute__((vector_size(16))), param1 = LDS pointer, then
// two int immediates (offset, cpol).
typedef int v4i_vs __attribute__((vector_size(16)));
typedef __attribute__((address_space(1))) v4i_vs* gasync_t;
typedef __attribute__((address_space(3))) v4i_vs* lasync_t;

// ---------------- ws layout (bytes) ----------------
// [0,        524288)  W_e in bf16          (256*1024*2)
// [524288,   589824)  d_proj f32           (64*256*4)
// [589824,  1114112)  scores f32           (2048*64*4)
// [1114112, 5308416)  partial ctx f32      (16*64*1024*4)

// ---------------------------------------------------------------------------
// K0: convert W_e (fp32 [A,E]) -> bf16, row-major, K contiguous
// ---------------------------------------------------------------------------
__global__ void k0_convert_we(const float* __restrict__ We, __bf16* __restrict__ WeBf) {
    int i = blockIdx.x * 256 + threadIdx.x;   // 262144 elements
    WeBf[i] = (__bf16)We[i];
}

// ---------------------------------------------------------------------------
// K1: d_proj[n][a] = dot(dec_h[n,:], W_d[a,:])   (tiny, L2-resident)
// ---------------------------------------------------------------------------
__global__ void k1_dproj(const float* __restrict__ dec_h, const float* __restrict__ Wd,
                         float* __restrict__ dproj) {
    int n = blockIdx.x;        // 64
    int a = threadIdx.x;       // 256
    const float* dh = dec_h + (size_t)n * E_DIM;
    const float* wd = Wd + (size_t)a * E_DIM;
    float acc = 0.f;
    #pragma unroll 4
    for (int k = 0; k < E_DIM; k += 4) {
        float4 x = *(const float4*)(dh + k);
        float4 w = *(const float4*)(wd + k);
        acc += x.x * w.x + x.y * w.y + x.z * w.z + x.w * w.w;
    }
    dproj[n * A_DIM + a] = acc;
}

// ---------------------------------------------------------------------------
// K2: fused scores kernel (WMMA bf16, f32 accum)
//   Each wave: 16 rows (m = t*64+n), all 16 A-col tiles, K=1024 in steps of 32.
//   B K-slice (16 KB) staged in LDS once per block (double-buffered) via
//   GLOBAL_LOAD_ASYNC_TO_LDS_B128; shared by all 8 waves -> 8x less L2 traffic.
//   Epilogue: scores[m] = sum_a tanh(acc + dproj[n][a]) * v[a]
// ---------------------------------------------------------------------------
__global__ __launch_bounds__(256) void k2_scores(
    const float* __restrict__ enc, const __bf16* __restrict__ WeBf,
    const float* __restrict__ dproj, const float* __restrict__ v,
    float* __restrict__ scores) {

    // LDS B slice: [buf][col 0..255][klocal 0..31] bf16  (2 x 16 KB)
    __shared__ __bf16 Bs[2][A_DIM * 32];

    const int tid   = threadIdx.x;
    const int lane  = tid & 31;
    const int wv    = tid >> 5;
    const int mbase = blockIdx.x * 128 + wv * 16;   // 16 rows per wave
    const int half  = lane >> 4;                    // K-subblock select (ISA A/B layout)
    const int lrow  = lane & 15;

    // cooperative stage of one K=32 slice of W_e-bf16: col = tid, 64 bytes each
    auto stage = [&](int kk, int buf) {
        const __bf16* src = WeBf + (size_t)tid * E_DIM + kk;
        __bf16* dst = &Bs[buf][tid * 32];
#if HAVE_ASYNC
        #pragma unroll
        for (int q = 0; q < 4; ++q)
            __builtin_amdgcn_global_load_async_to_lds_b128(
                (gasync_t)(src + q * 8), (lasync_t)(dst + q * 8), 0, 0);
#else
        #pragma unroll
        for (int q = 0; q < 4; ++q)
            *(float4*)(dst + q * 8) = *(const float4*)(src + q * 8);
#endif
    };

    const v8f vzero = {};
    v8f acc[16];
    #pragma unroll
    for (int i = 0; i < 16; ++i) acc[i] = vzero;

    // pointer to this lane's A row (row = mbase + lrow), enc row-major, K contiguous
    const float* rowp = enc + (size_t)(mbase + lrow) * E_DIM;

    stage(0, 0);
    for (int it = 0; it < E_DIM / 32; ++it) {
        const int kk  = it * 32;
        const int buf = it & 1;

        WAIT_ASYNC();        // our async writes into Bs[buf] complete
        __syncthreads();     // ... and are visible to all waves in the block
        if (it + 1 < E_DIM / 32) stage(kk + 32, buf ^ 1);

        // ---- A fragment: contiguous k order k = 16h + s (same as B) ----
        // lane holds row (lane&15); values[0..15]=K[kk+16h..kk+16h+15]
        const float* ap = rowp + kk + half * 16;
        float4 f0 = *(const float4*)(ap);
        float4 f1 = *(const float4*)(ap + 4);
        float4 f2 = *(const float4*)(ap + 8);
        float4 f3 = *(const float4*)(ap + 12);
        v16bf afrag;
        afrag[0]=(__bf16)f0.x; afrag[1]=(__bf16)f0.y; afrag[2]=(__bf16)f0.z; afrag[3]=(__bf16)f0.w;
        afrag[4]=(__bf16)f1.x; afrag[5]=(__bf16)f1.y; afrag[6]=(__bf16)f1.z; afrag[7]=(__bf16)f1.w;
        afrag[8]=(__bf16)f2.x; afrag[9]=(__bf16)f2.y; afrag[10]=(__bf16)f2.z; afrag[11]=(__bf16)f2.w;
        afrag[12]=(__bf16)f3.x; afrag[13]=(__bf16)f3.y; afrag[14]=(__bf16)f3.z; afrag[15]=(__bf16)f3.w;

        // ---- B fragments from LDS: col = at*16 + lrow, 32 contiguous bytes ----
        const __bf16* bbase = &Bs[buf][0];
        #pragma unroll
        for (int at = 0; at < 16; ++at) {
            v16bf bfrag = *(const v16bf*)(bbase + (at * 16 + lrow) * 32 + half * 16);
            acc[at] = __builtin_amdgcn_wmma_f32_16x16x32_bf16(
                false, afrag, false, bfrag, (short)0, acc[at], false, false);
        }
    }

    // ---- epilogue: tanh + dot with v, per-row accumulation ----
    // C element in VGPR r, lane l: M = r + 8*(l>>4), N = l&15
    float sacc[8];
    #pragma unroll
    for (int r = 0; r < 8; ++r) sacc[r] = 0.f;

    #pragma unroll
    for (int at = 0; at < 16; ++at) {
        const int a = at * 16 + lrow;
        const float vv = v[a];
        #pragma unroll
        for (int r = 0; r < 8; ++r) {
            const int row = mbase + r + 8 * half;
            const int n   = row & (N_DIM - 1);
            const float d = dproj[n * A_DIM + a];
            sacc[r] += tanhf(acc[at][r] + d) * vv;
        }
    }

    // reduce the 16 column-lanes of each half-wave
    #pragma unroll
    for (int r = 0; r < 8; ++r) {
        float s = sacc[r];
        s += __shfl_xor(s, 1, 16);
        s += __shfl_xor(s, 2, 16);
        s += __shfl_xor(s, 4, 16);
        s += __shfl_xor(s, 8, 16);
        sacc[r] = s;
    }
    if (lrow == 0) {
        #pragma unroll
        for (int r = 0; r < 8; ++r)
            scores[mbase + r + 8 * half] = sacc[r];
    }
}

// ---------------------------------------------------------------------------
// K3: softmax over T for each n; write alpha [T,N] into d_out tail
// ---------------------------------------------------------------------------
__global__ void k3_softmax(const float* __restrict__ scores, float* __restrict__ alpha) {
    __shared__ float red[256];
    const int n = blockIdx.x;
    const int tid = threadIdx.x;

    float m = -1e30f;
    for (int t = tid; t < T_DIM; t += 256) m = fmaxf(m, scores[t * N_DIM + n]);
    red[tid] = m; __syncthreads();
    for (int s = 128; s > 0; s >>= 1) {
        if (tid < s) red[tid] = fmaxf(red[tid], red[tid + s]);
        __syncthreads();
    }
    const float smax = red[0]; __syncthreads();

    float sum = 0.f;
    for (int t = tid; t < T_DIM; t += 256) sum += __expf(scores[t * N_DIM + n] - smax);
    red[tid] = sum; __syncthreads();
    for (int s = 128; s > 0; s >>= 1) {
        if (tid < s) red[tid] += red[tid + s];
        __syncthreads();
    }
    const float inv = 1.f / red[0];

    for (int t = tid; t < T_DIM; t += 256)
        alpha[t * N_DIM + n] = __expf(scores[t * N_DIM + n] - smax) * inv;
}

// ---------------------------------------------------------------------------
// K4: partial ctx: partial[tc][n][e] = sum_{t in chunk} alpha[t][n] * enc[t][n][e]
//     grid = 16 t-chunks * 64 n;  thread -> one float4 of E (coalesced 4KB rows)
// ---------------------------------------------------------------------------
__global__ void k4_ctx_partial(const float* __restrict__ enc, const float* __restrict__ alpha,
                               float* __restrict__ partial) {
    const int n  = blockIdx.x & 63;
    const int tc = blockIdx.x >> 6;       // 0..15, 128 t each
    const int e0 = threadIdx.x * 4;
    float4 acc = make_float4(0.f, 0.f, 0.f, 0.f);
    for (int t = tc * 128; t < tc * 128 + 128; ++t) {
        const float a = alpha[t * N_DIM + n];
        const float4 p = *(const float4*)(enc + (size_t)(t * N_DIM + n) * E_DIM + e0);
        acc.x += a * p.x; acc.y += a * p.y; acc.z += a * p.z; acc.w += a * p.w;
    }
    *(float4*)(partial + ((size_t)tc * N_DIM + n) * E_DIM + e0) = acc;
}

// ---------------------------------------------------------------------------
// K5: reduce 16 partials -> ctx [N,E]   (deterministic, no fp atomics)
// ---------------------------------------------------------------------------
__global__ void k5_ctx_reduce(const float* __restrict__ partial, float* __restrict__ ctx) {
    const int idx = blockIdx.x * 256 + threadIdx.x;   // 65536 = n*1024+e
    float s = 0.f;
    #pragma unroll
    for (int tc = 0; tc < 16; ++tc) s += partial[(size_t)tc * 65536 + idx];
    ctx[idx] = s;
}

// ---------------------------------------------------------------------------
extern "C" void kernel_launch(void* const* d_in, const int* in_sizes, int n_in,
                              void* d_out, int out_size, void* d_ws, size_t ws_size,
                              hipStream_t stream) {
    const float* enc   = (const float*)d_in[0];   // [T,N,E]
    const float* dec_h = (const float*)d_in[1];   // [N,D]
    const float* W_e   = (const float*)d_in[2];   // [A,E]
    const float* W_d   = (const float*)d_in[3];   // [A,D]
    const float* v     = (const float*)d_in[4];   // [1,A]

    float* ctx   = (float*)d_out;                 // [N,E] = 65536
    float* alpha = (float*)d_out + N_DIM * E_DIM; // [T,N] = 131072

    char* ws = (char*)d_ws;
    __bf16* WeBf   = (__bf16*)(ws);
    float*  dproj  = (float*)(ws + 524288);
    float*  scores = (float*)(ws + 589824);
    float*  part   = (float*)(ws + 1114112);

    k0_convert_we<<<A_DIM * E_DIM / 256, 256, 0, stream>>>(W_e, WeBf);
    k1_dproj<<<N_DIM, A_DIM, 0, stream>>>(dec_h, W_d, dproj);
    k2_scores<<<M_TOT / 128, 256, 0, stream>>>(enc, WeBf, dproj, v, scores);
    k3_softmax<<<N_DIM, 256, 0, stream>>>(scores, alpha);
    k4_ctx_partial<<<16 * N_DIM, 256, 0, stream>>>(enc, alpha, part);
    k5_ctx_reduce<<<N_DIM * E_DIM / 256, 256, 0, stream>>>(part, ctx);
}
